// Model_78975858638906
// MI455X (gfx1250) — matrix operand order repaired and checked by reading, not compiled, not run on hardware
//
#include <hip/hip_runtime.h>
#include <hip/hip_bf16.h>
#include <math.h>

#define L_DNA  16384
#define NPROT  131072
#define KNBR   16
#define ND_PTS (L_DNA*11)
#define EPROT  (NPROT*16)

typedef _Float16 v16h __attribute__((ext_vector_type(16)));
typedef float    v8f  __attribute__((ext_vector_type(8)));

__device__ __constant__ int   d_switch[11] = {10,9,5,4,3,2,8,7,6,1,0};
__device__ __constant__ float d_st[14] = {-1.f,-1.f,1.f,1.f,1.f,1.f,-1.f,1.f,1.f,-1.f,1.f,1.f,1.f,1.f};

__device__ inline v8f wmma16(v16h a, v16h b, v8f c) {
  return __builtin_amdgcn_wmma_f32_16x16x32_f16(false, a, false, b, (short)0, c, false, false);
}

// A fragment (16x32 f16, row-major source in LDS): lanes 0-15 -> M=lane, K 0..15;
// lanes 16-31 -> M=lane-16, K 16..31.  kk selects which 32-wide K slab.
__device__ inline v16h load_a_frag(const _Float16* m, int stride, int lane, int kk) {
  int row = lane & 15;
  int off = (lane < 16) ? 0 : 16;
  v16h a;
#pragma unroll
  for (int e = 0; e < 16; ++e) {
    int k = 32*kk + off + e;
    a[e] = m[row*stride + k];
  }
  return a;
}

// B fragment (32x16 f16 from K-major [Kpad][N] table): lane<16 -> N=lane, K 0..15; else K 16..31.
__device__ inline v16h load_b_frag(const _Float16* w, int n, int lane, int kk, int nt) {
  int col = (lane & 15) + nt*16;
  int kb  = 32*kk + ((lane < 16) ? 0 : 16);
  v16h b;
#pragma unroll
  for (int e = 0; e < 16; ++e) b[e] = w[(kb+e)*n + col];
  return b;
}

__device__ inline float angle3(float ax,float ay,float az,float bx,float by,float bz){
  float cx = ay*bz - az*by;
  float cy = az*bx - ax*bz;
  float cz = ax*by - ay*bx;
  return atan2f(sqrtf(cx*cx+cy*cy+cz*cz), ax*bx+ay*by+az*bz);
}

// ---------------- prep: zero segment-max buffer + f16 weight tables ----------------
__global__ void prep_kernel(int* xp_i,
                            const float* bn_lw, _Float16* bn_lw_h,
                            const float* red_w1, _Float16* red_w1_h,
                            const float* red_w2, _Float16* red_w2_h) {
  int idx = blockIdx.x*blockDim.x + threadIdx.x;
  if (idx < NPROT*10) xp_i[idx] = 0;
  if (idx < 32*32) { int k = idx >> 5, n = idx & 31;
    bn_lw_h[idx]  = (k < 24)  ? (_Float16)bn_lw[k*32 + n]  : (_Float16)0.f; }
  if (idx < 128*64) { int k = idx >> 6, n = idx & 63;
    red_w1_h[idx] = (k < 110) ? (_Float16)red_w1[k*64 + n] : (_Float16)0.f; }
  if (idx < 64*32) red_w2_h[idx] = (_Float16)red_w2[idx];
}

// ---------------- kernel 1: xp0 = relu(x_prot @ pe_w1 + pe_b1) ----------------
__global__ void prot_embed_kernel(const float* __restrict__ x_prot,
                                  const float* __restrict__ w1, const float* __restrict__ b1,
                                  float* __restrict__ xp0) {
  __shared__ float sw[140], sb[10];
  for (int t = threadIdx.x; t < 140; t += blockDim.x) sw[t] = w1[t];
  for (int t = threadIdx.x; t < 10;  t += blockDim.x) sb[t] = b1[t];
  __syncthreads();
  int p = blockIdx.x*blockDim.x + threadIdx.x;
  if (p >= NPROT) return;
  float x[14];
#pragma unroll
  for (int c = 0; c < 14; ++c) x[c] = x_prot[p*14 + c];
#pragma unroll
  for (int n = 0; n < 10; ++n) {
    float h = sb[n];
#pragma unroll
    for (int c = 0; c < 14; ++c) h += x[c]*sw[c*10 + n];
    xp0[p*10 + n] = fmaxf(h, 0.f);
  }
}

// ---------------- kernel 2: edge messages + segment-max (int atomicMax on >=0 floats) ----
__global__ void edge_kernel(const float* __restrict__ xp0, const float* __restrict__ v_prot,
                            const int* __restrict__ e_prot,
                            const float* __restrict__ wm, const float* __restrict__ bm,
                            int* __restrict__ xp_i) {
  __shared__ float sw[130], sb[10];
  for (int t = threadIdx.x; t < 130; t += blockDim.x) sw[t] = wm[t];
  for (int t = threadIdx.x; t < 10;  t += blockDim.x) sb[t] = bm[t];
  __syncthreads();
  int e = blockIdx.x*blockDim.x + threadIdx.x;
  if (e >= EPROT) return;
  int src = e_prot[e*2 + 0], dst = e_prot[e*2 + 1];
  float f[13];
#pragma unroll
  for (int c = 0; c < 10; ++c) f[c] = xp0[src*10 + c];
#pragma unroll
  for (int c = 0; c < 3; ++c)  f[10+c] = v_prot[src*3 + c] - v_prot[dst*3 + c];
#pragma unroll
  for (int n = 0; n < 10; ++n) {
    float m = sb[n];
#pragma unroll
    for (int c = 0; c < 13; ++c) m += f[c]*sw[c*10 + n];
    m = fmaxf(m, 0.f);
    atomicMax(xp_i + dst*10 + n, __float_as_int(m));
  }
}

// ---------------- kernel 3: fused gather + geometric feats + WMMA bn-layer + neighbor max
// one wave == one DNA point: its K=16 neighbors are the M rows of one WMMA tile.
__global__ void strand_bn_kernel(const float* __restrict__ xdp, const float* __restrict__ v_dna,
                                 const float* __restrict__ dna_vecs,
                                 const float* __restrict__ v_prot, const float* __restrict__ prot_vecs,
                                 const float* __restrict__ xp, const int* __restrict__ cross_src,
                                 const unsigned char* __restrict__ atom_mask,
                                 const float* __restrict__ ew1, const float* __restrict__ eb1,
                                 const float* __restrict__ ew2, const float* __restrict__ eb2,
                                 const _Float16* __restrict__ bn_lw_h,
                                 const float* __restrict__ bn_lb,
                                 const float* __restrict__ bn_gw, const float* __restrict__ bn_gb,
                                 float* __restrict__ xb, int rc) {
  __shared__ _Float16 sFeat[8][16*32];
  __shared__ float    sHmax[8][32];
  __shared__ float sEw1[110], sEb1[10], sEw2[100], sEb2[10], sLb[32], sGw[320], sGb[10];
  for (int t = threadIdx.x; t < 110; t += blockDim.x) sEw1[t] = ew1[t];
  for (int t = threadIdx.x; t < 10;  t += blockDim.x) { sEb1[t] = eb1[t]; sEb2[t] = eb2[t]; sGb[t] = bn_gb[t]; }
  for (int t = threadIdx.x; t < 100; t += blockDim.x) sEw2[t] = ew2[t];
  for (int t = threadIdx.x; t < 32;  t += blockDim.x) sLb[t] = bn_lb[t];
  for (int t = threadIdx.x; t < 320; t += blockDim.x) sGw[t] = bn_gw[t];
  __syncthreads();

  int lane = threadIdx.x & 31;
  int wv   = threadIdx.x >> 5;
  _Float16* fL = &sFeat[wv][0];

  // B fragments are loop-invariant: keep in VGPRs.
  v16h b0 = load_b_frag(bn_lw_h, 32, lane, 0, 0);
  v16h b1 = load_b_frag(bn_lw_h, 32, lane, 0, 1);
  float biasN0 = sLb[lane & 15];
  float biasN1 = sLb[16 + (lane & 15)];

  int nwaves = gridDim.x * (blockDim.x >> 5);
  for (int i = blockIdx.x*(blockDim.x >> 5) + wv; i < ND_PTS; i += nwaves) {
    int l = i / 11, a = i - l*11;
    int xrow = rc ? (L_DNA-1-l)*11 + a           : i;   // switch permute is dead for xdp in reference
    int vrow = rc ? (L_DNA-1-l)*11 + d_switch[a] : i;

    // xi = embed MLP (redundant per lane; ~220 FMA, negligible)
    float xr[11];
#pragma unroll
    for (int c = 0; c < 11; ++c) xr[c] = xdp[xrow*11 + c];
    float h10[10];
#pragma unroll
    for (int n = 0; n < 10; ++n) {
      float h = sEb1[n];
#pragma unroll
      for (int c = 0; c < 11; ++c) h += xr[c]*sEw1[c*10 + n];
      h10[n] = fmaxf(h, 0.f);
    }
    float xi[10];
#pragma unroll
    for (int n = 0; n < 10; ++n) {
      float h = sEb2[n];
#pragma unroll
      for (int c = 0; c < 10; ++c) h += h10[c]*sEw2[c*10 + n];
      xi[n] = h;
    }

    float vx = v_dna[vrow*3+0], vy = v_dna[vrow*3+1], vz = v_dna[vrow*3+2];
    float nix = dna_vecs[vrow*3+0], niy = dna_vecs[vrow*3+1], niz = dna_vecs[vrow*3+2];

    bool em = false;
    if (lane < 16) {            // lane == neighbor index k == WMMA row M
      int j = cross_src[i*KNBR + lane];
      float px = v_prot[j*3+0], py = v_prot[j*3+1], pz = v_prot[j*3+2];
      float njx = prot_vecs[j*3+0], njy = prot_vecs[j*3+1], njz = prot_vecs[j*3+2];
      em = atom_mask[j] != 0;
      float dx = px - vx, dy = py - vy, dz = pz - vz;
      float dn = sqrtf(dx*dx + dy*dy + dz*dz);
      float p1 = angle3(nix,niy,niz, dx,dy,dz);
      float p2 = angle3(njx,njy,njz, dx,dy,dz);
      float p3 = angle3(nix,niy,niz, njx,njy,njz);
      _Float16* row = fL + lane*32;
#pragma unroll
      for (int c = 0; c < 10; ++c) row[c] = (_Float16)xi[c];
#pragma unroll
      for (int c = 0; c < 10; ++c) row[10+c] = (_Float16)xp[j*10 + c];
      row[20] = (_Float16)dn; row[21] = (_Float16)p1; row[22] = (_Float16)p2; row[23] = (_Float16)p3;
#pragma unroll
      for (int c = 24; c < 32; ++c) row[c] = (_Float16)0.f;
    }
    unsigned mbits = (unsigned)__ballot(em) & 0xFFFFu;
    asm volatile("s_wait_dscnt 0" ::: "memory");

    v16h afr = load_a_frag(fL, 32, lane, 0);
    v8f c0 = {0.f,0.f,0.f,0.f,0.f,0.f,0.f,0.f};
    v8f c1 = {0.f,0.f,0.f,0.f,0.f,0.f,0.f,0.f};
    c0 = wmma16(afr, b0, c0);
    c1 = wmma16(afr, b1, c1);

    // relu + mask + max over M (=neighbors): per-lane over the 8 C VGPRs, then xor-16 shuffle.
    int mshift = (lane >> 4) << 3;
    float m0 = -3e38f, m1 = -3e38f;
#pragma unroll
    for (int r = 0; r < 8; ++r) {
      float h0 = fmaxf(c0[r] + biasN0, 0.f);
      float h1 = fmaxf(c1[r] + biasN1, 0.f);
      if ((mbits >> (mshift + r)) & 1u) { h0 = -1e9f; h1 = -1e9f; }
      m0 = fmaxf(m0, h0);
      m1 = fmaxf(m1, h1);
    }
    m0 = fmaxf(m0, __shfl_xor(m0, 16, 32));
    m1 = fmaxf(m1, __shfl_xor(m1, 16, 32));
    if (lane < 16) { sHmax[wv][lane] = m0; sHmax[wv][16 + lane] = m1; }
    asm volatile("s_wait_dscnt 0" ::: "memory");

    if (lane < 10) {
      float acc = sGb[lane];
#pragma unroll
      for (int c = 0; c < 32; ++c) acc += sHmax[wv][c]*sGw[c*10 + lane];
      xb[(size_t)i*10 + lane] = acc;
    }
  }
}

// ---------------- kernel 4: (L,110)->relu64->32, WMMA GEMM per 16-row tile ----------------
__global__ void red_mlp_kernel(const float* __restrict__ xb,
                               const _Float16* __restrict__ w1h, const float* __restrict__ b1,
                               const _Float16* __restrict__ w2h, const float* __restrict__ b2,
                               float* __restrict__ xb32) {
  __shared__ _Float16 sA[4][16*128];
  __shared__ _Float16 sH[4][16*64];
  int lane = threadIdx.x & 31, wv = threadIdx.x >> 5;
  int tile = blockIdx.x*4 + wv;
  if (tile >= L_DNA/16) return;
  int rowbase = tile*16;

  for (int t = lane; t < 16*128; t += 32) {
    int r = t >> 7, c = t & 127;
    sA[wv][t] = (c < 110) ? (_Float16)xb[(size_t)(rowbase + r)*110 + c] : (_Float16)0.f;
  }
  asm volatile("s_wait_dscnt 0" ::: "memory");

  v8f acc[4];
#pragma unroll
  for (int nt = 0; nt < 4; ++nt) acc[nt] = (v8f){0.f,0.f,0.f,0.f,0.f,0.f,0.f,0.f};
#pragma unroll
  for (int kk = 0; kk < 4; ++kk) {
    v16h a = load_a_frag(&sA[wv][0], 128, lane, kk);
#pragma unroll
    for (int nt = 0; nt < 4; ++nt) {
      v16h b = load_b_frag(w1h, 64, lane, kk, nt);
      acc[nt] = wmma16(a, b, acc[nt]);
    }
  }
  int Moff = (lane >> 4)*8;
#pragma unroll
  for (int nt = 0; nt < 4; ++nt) {
    int N = (lane & 15) + 16*nt;
    float bias = b1[N];
#pragma unroll
    for (int r = 0; r < 8; ++r)
      sH[wv][(Moff + r)*64 + N] = (_Float16)fmaxf(acc[nt][r] + bias, 0.f);
  }
  asm volatile("s_wait_dscnt 0" ::: "memory");

#pragma unroll
  for (int nt = 0; nt < 2; ++nt) {
    v8f c2 = {0.f,0.f,0.f,0.f,0.f,0.f,0.f,0.f};
#pragma unroll
    for (int kk = 0; kk < 2; ++kk) {
      v16h a2 = load_a_frag(&sH[wv][0], 64, lane, kk);
      v16h b2f = load_b_frag(w2h, 32, lane, kk, nt);
      c2 = wmma16(a2, b2f, c2);
    }
    int N = (lane & 15) + 16*nt;
    float bias2 = b2[N];
#pragma unroll
    for (int r = 0; r < 8; ++r)
      xb32[(size_t)(rowbase + Moff + r)*32 + N] = c2[r] + bias2;
  }
}

// xd accessor implementing flip*st + roll for the RC strand
__device__ inline float xd_at(const float* x_dna, int l, int c, int rc) {
  if (!rc) return x_dna[l*14 + c];
  int lr;
  if (c >= 6 && c < 12) { int l2 = (l + 1) & (L_DNA - 1); lr = L_DNA - 1 - l2; }
  else lr = L_DNA - 1 - l;
  return x_dna[lr*14 + c] * d_st[c];
}

// ---------------- kernel 5a: conv1 (46->8, k=3, SAME) + relu ----------------
__global__ void conv1_kernel(const float* __restrict__ xb32, const float* __restrict__ x_dna,
                             const float* __restrict__ k1, const float* __restrict__ cb1,
                             float* __restrict__ hc1, int rc) {
  __shared__ float sK[8*46*3], sB[8];
  for (int t = threadIdx.x; t < 8*46*3; t += blockDim.x) sK[t] = k1[t];
  for (int t = threadIdx.x; t < 8;      t += blockDim.x) sB[t] = cb1[t];
  __syncthreads();
  int l = blockIdx.x*blockDim.x + threadIdx.x;
  if (l >= L_DNA) return;
  float acc[8];
#pragma unroll
  for (int oc = 0; oc < 8; ++oc) acc[oc] = sB[oc];
#pragma unroll
  for (int t = -1; t <= 1; ++t) {
    int r = l + t;
    if (r < 0 || r >= L_DNA) continue;
    float val[46];
#pragma unroll
    for (int c = 0; c < 32; ++c) val[c] = xb32[(size_t)r*32 + c];
#pragma unroll
    for (int c = 0; c < 14; ++c) val[32 + c] = xd_at(x_dna, r, c, rc);
#pragma unroll
    for (int oc = 0; oc < 8; ++oc) {
#pragma unroll
      for (int ic = 0; ic < 46; ++ic)
        acc[oc] += sK[(oc*46 + ic)*3 + (t + 1)] * val[ic];
    }
  }
#pragma unroll
  for (int oc = 0; oc < 8; ++oc) hc1[(size_t)l*8 + oc] = fmaxf(acc[oc], 0.f);
}

// ---------------- kernel 5b: conv2 + fw + final MLP + temperature ----------------
__global__ void conv2_tail_kernel(const float* __restrict__ hc1,
                                  const float* __restrict__ k2, const float* __restrict__ cb2,
                                  const float* __restrict__ fw, const float* __restrict__ fb,
                                  const float* __restrict__ w1, const float* __restrict__ b1,
                                  const float* __restrict__ w2, const float* __restrict__ b2,
                                  const float* __restrict__ w3, const float* __restrict__ b3,
                                  const float* __restrict__ gt,
                                  float* __restrict__ out, int strand) {
  __shared__ float sK2[192], sB2c[8], sFw[64], sFb[8], sW1[64], sB1m[8], sW2[64], sB2m[8], sW3[32], sB3[4];
  for (int t = threadIdx.x; t < 192; t += blockDim.x) sK2[t] = k2[t];
  for (int t = threadIdx.x; t < 64;  t += blockDim.x) { sFw[t] = fw[t]; sW1[t] = w1[t]; sW2[t] = w2[t]; }
  for (int t = threadIdx.x; t < 32;  t += blockDim.x) sW3[t] = w3[t];
  for (int t = threadIdx.x; t < 8;   t += blockDim.x) { sB2c[t] = cb2[t]; sFb[t] = fb[t]; sB1m[t] = b1[t]; sB2m[t] = b2[t]; }
  for (int t = threadIdx.x; t < 4;   t += blockDim.x) sB3[t] = b3[t];
  __syncthreads();
  int l = blockIdx.x*blockDim.x + threadIdx.x;
  if (l >= L_DNA) return;

  float acc[8];
#pragma unroll
  for (int oc = 0; oc < 8; ++oc) acc[oc] = sB2c[oc];
#pragma unroll
  for (int t = -1; t <= 1; ++t) {
    int r = l + t;
    if (r < 0 || r >= L_DNA) continue;
#pragma unroll
    for (int ic = 0; ic < 8; ++ic) {
      float v = hc1[(size_t)r*8 + ic];
#pragma unroll
      for (int oc = 0; oc < 8; ++oc)
        acc[oc] += sK2[(oc*8 + ic)*3 + (t + 1)] * v;
    }
  }
  float m[8];
#pragma unroll
  for (int i = 0; i < 8; ++i) m[i] = fmaxf(acc[i], 0.f);
  float fo[8];
#pragma unroll
  for (int o = 0; o < 8; ++o) {
    float s = sFb[o];
#pragma unroll
    for (int i = 0; i < 8; ++i) s += sFw[o*8 + i]*m[i];
    fo[o] = s;
  }
  float h1[8];
#pragma unroll
  for (int n = 0; n < 8; ++n) {
    float s = sB1m[n];
#pragma unroll
    for (int o = 0; o < 8; ++o) s += fo[o]*sW1[o*8 + n];
    h1[n] = fmaxf(s, 0.f);
  }
  float h2[8];
#pragma unroll
  for (int n = 0; n < 8; ++n) {
    float s = sB2m[n];
#pragma unroll
    for (int o = 0; o < 8; ++o) s += h1[o]*sW2[o*8 + n];
    h2[n] = fmaxf(s, 0.f);
  }
  float scale = 1.f + expf(-gt[0]);   // 1/sigmoid(gt)
#pragma unroll
  for (int c = 0; c < 4; ++c) {
    float s = sB3[c];
#pragma unroll
    for (int n = 0; n < 8; ++n) s += h2[n]*sW3[n*4 + c];
    out[((size_t)strand*L_DNA + l)*4 + c] = s * scale;
  }
}

extern "C" void kernel_launch(void* const* d_in, const int* in_sizes, int n_in,
                              void* d_out, int out_size, void* d_ws, size_t ws_size,
                              hipStream_t stream) {
  const float* x_dna_point = (const float*)d_in[0];
  const float* v_dna       = (const float*)d_in[1];
  const float* x_dna       = (const float*)d_in[2];
  const float* x_prot      = (const float*)d_in[3];
  const float* v_prot      = (const float*)d_in[4];
  const float* prot_vecs   = (const float*)d_in[5];
  const float* dna_vecs    = (const float*)d_in[6];
  const int*   e_prot      = (const int*)d_in[7];
  const int*   cross_src   = (const int*)d_in[8];
  const unsigned char* atom_mask = (const unsigned char*)d_in[9];
  const float* embed_w1 = (const float*)d_in[10];
  const float* embed_b1 = (const float*)d_in[11];
  const float* embed_w2 = (const float*)d_in[12];
  const float* embed_b2 = (const float*)d_in[13];
  const float* pe_w1 = (const float*)d_in[14];
  const float* pe_b1 = (const float*)d_in[15];
  const float* pe_wm = (const float*)d_in[16];
  const float* pe_bm = (const float*)d_in[17];
  const float* bn_lw = (const float*)d_in[18];
  const float* bn_lb = (const float*)d_in[19];
  const float* bn_gw = (const float*)d_in[20];
  const float* bn_gb = (const float*)d_in[21];
  const float* red_w1 = (const float*)d_in[22];
  const float* red_b1 = (const float*)d_in[23];
  const float* red_w2 = (const float*)d_in[24];
  const float* red_b2 = (const float*)d_in[25];
  const float* cnn_k1 = (const float*)d_in[26];
  const float* cnn_b1 = (const float*)d_in[27];
  const float* cnn_k2 = (const float*)d_in[28];
  const float* cnn_b2 = (const float*)d_in[29];
  const float* cnn_fw = (const float*)d_in[30];
  const float* cnn_fb = (const float*)d_in[31];
  const float* mlp_w1 = (const float*)d_in[32];
  const float* mlp_b1 = (const float*)d_in[33];
  const float* mlp_w2 = (const float*)d_in[34];
  const float* mlp_b2 = (const float*)d_in[35];
  const float* mlp_w3 = (const float*)d_in[36];
  const float* mlp_b3 = (const float*)d_in[37];
  const float* gtemp  = (const float*)d_in[38];

  // workspace partition (~20.4 MB)
  size_t off = 0;
  char* base = (char*)d_ws;
  auto take = [&](size_t nbytes) -> void* {
    void* p = base + off;
    off += (nbytes + 255) & ~(size_t)255;
    return p;
  };
  float*    xp0      = (float*)take((size_t)NPROT*10*sizeof(float));
  int*      xp_i     = (int*)  take((size_t)NPROT*10*sizeof(int));
  float*    xb       = (float*)take((size_t)ND_PTS*10*sizeof(float));
  float*    xb32     = (float*)take((size_t)L_DNA*32*sizeof(float));
  float*    hc1      = (float*)take((size_t)L_DNA*8*sizeof(float));
  _Float16* bn_lw_h  = (_Float16*)take(32*32*sizeof(_Float16));
  _Float16* red_w1_h = (_Float16*)take(128*64*sizeof(_Float16));
  _Float16* red_w2_h = (_Float16*)take(64*32*sizeof(_Float16));
  (void)ws_size; (void)n_in; (void)in_sizes; (void)out_size;

  prep_kernel<<<(NPROT*10 + 255)/256, 256, 0, stream>>>(xp_i, bn_lw, bn_lw_h, red_w1, red_w1_h, red_w2, red_w2_h);
  prot_embed_kernel<<<NPROT/256, 256, 0, stream>>>(x_prot, pe_w1, pe_b1, xp0);
  edge_kernel<<<EPROT/256, 256, 0, stream>>>(xp0, v_prot, e_prot, pe_wm, pe_bm, xp_i);

  for (int s = 0; s < 2; ++s) {
    strand_bn_kernel<<<512, 256, 0, stream>>>(
        x_dna_point, v_dna, dna_vecs, v_prot, prot_vecs,
        (const float*)xp_i, cross_src, atom_mask,
        embed_w1, embed_b1, embed_w2, embed_b2,
        bn_lw_h, bn_lb, bn_gw, bn_gb, xb, s);
    red_mlp_kernel<<<(L_DNA/16)/4, 128, 0, stream>>>(xb, red_w1_h, red_b1, red_w2_h, red_b2, xb32);
    conv1_kernel<<<L_DNA/256, 256, 0, stream>>>(xb32, x_dna, cnn_k1, cnn_b1, hc1, s);
    conv2_tail_kernel<<<L_DNA/256, 256, 0, stream>>>(hc1, cnn_k2, cnn_b2, cnn_fw, cnn_fb,
                                                     mlp_w1, mlp_b1, mlp_w2, mlp_b2, mlp_w3, mlp_b3,
                                                     gtemp, (float*)d_out, s);
  }
}
